// UIMixFormer_86998857548019
// MI455X (gfx1250) — compile-verified, matrix operand order checked by reading, not run on hardware
//
#include <hip/hip_runtime.h>
#include <math.h>

// ---------------- model dims ----------------
#define B_   512
#define T_   200
#define N_   8
#define D_   64
#define NU_  4
#define H_   256
#define SH_  1024
#define TD_  512
#define L_   4
#define BT_  (B_ * T_)
#define EPS_ 1e-6f

// ---------------- bf16 helpers (native converts) ----------------
typedef __attribute__((ext_vector_type(16))) __bf16       bf16x16;
typedef __attribute__((ext_vector_type(8)))  float        floatx8;
typedef __attribute__((ext_vector_type(4)))  unsigned int uintx4;
typedef __attribute__((ext_vector_type(4)))  int          intx4;

__device__ __forceinline__ float bf2f(unsigned short h) {
    return (float)__builtin_bit_cast(__bf16, h);
}
__device__ __forceinline__ unsigned short f2bf(float f) {
    __bf16 h = (__bf16)f;
    return __builtin_bit_cast(unsigned short, h);
}

union FragU { bf16x16 v; unsigned int u[8]; };
union AccU  { floatx8 v; float f[8]; };

// async global->LDS path (gfx1250); falls back to ds stores if builtin absent
#if defined(__has_builtin)
#if __has_builtin(__builtin_amdgcn_global_load_async_to_lds_b128)
#define ASYNC_LDS 1
#endif
#endif
#ifndef ASYNC_LDS
#define ASYNC_LDS 0
#endif
#if ASYNC_LDS
typedef __attribute__((address_space(3))) intx4 lds_intx4;
#define STAGE_CP(gsrc, ldst) \
    __builtin_amdgcn_global_load_async_to_lds_b128((intx4*)(gsrc), (lds_intx4*)(ldst), 0, 0)
#define STAGE_WAIT() asm volatile("s_wait_asynccnt 0x0" ::: "memory")
#else
#define STAGE_CP(gsrc, ldst) *(uintx4*)(ldst) = *(const uintx4*)(gsrc)
#define STAGE_WAIT()
#endif

// ---------------- generic batched bf16 WMMA GEMM ----------------
// C[M,N] = A[M,K] (bf16) x Bt[N,K]^T (bf16, pre-transposed weights), optional
// per-A-row f32 scale applied to C rows in the epilogue (diag(s)AB == diag(s)(AB)).
// 128 threads = 4 waves; block tile 64 x TN; wave tile 32 x (TN/2).
// LDS double-buffered; staging is pure lane-wise b128 copies (async-to-LDS capable).
// M mult of 64, N mult of TN, K mult of 32 (true at all call sites).
template<int TN>
__global__ __launch_bounds__(128)
void gemm_bf16_kernel(const unsigned short* __restrict__ A, int lda, long long sAz,
                      const float* __restrict__ rowScale,
                      const unsigned short* __restrict__ Bt, int ldb, long long sBz,
                      float* __restrict__ Cf, unsigned short* __restrict__ Cbf,
                      int ldc, long long sCz, int K)
{
    constexpr int JT = TN / 32;                 // 16-wide col tiles per wave (2 or 4)
    __shared__ unsigned short As[2][64][40];    // padded rows: 80B -> 16B-aligned b128
    __shared__ unsigned short Bs[2][TN][40];    // Bs[n][k], K-contiguous

    const int tid = threadIdx.x;
    A  += (long long)blockIdx.z * sAz;
    Bt += (long long)blockIdx.z * sBz;
    const long long cOff = (long long)blockIdx.z * sCz;
    const int mBase = blockIdx.y * 64;
    const int nBase = blockIdx.x * TN;
    const int lane  = tid & 31, wave = tid >> 5;
    const int wRow  = (wave >> 1) * 32;
    const int wCol  = (wave & 1) * (TN / 2);
    const int lm    = lane & 15, half = lane >> 4;

    AccU c[2][JT];
#pragma unroll
    for (int i = 0; i < 2; ++i)
#pragma unroll
        for (int j = 0; j < JT; ++j)
#pragma unroll
            for (int e = 0; e < 8; ++e) c[i][j].f[e] = 0.f;

    auto stage = [&](int buf, int kb) {
        // A tile 64x32: 256 lane-chunks of 8 elems
#pragma unroll
        for (int ch = 0; ch < 2; ++ch) {
            int e8 = tid + ch * 128;
            int ar = e8 >> 2, ac = (e8 & 3) * 8;
            STAGE_CP(A + (long long)(mBase + ar) * lda + kb + ac, &As[buf][ar][ac]);
        }
        // Bt tile TNx32 (K-contiguous rows): TN*4 lane-chunks of 8 elems
#pragma unroll
        for (int ch = 0; ch < TN / 32; ++ch) {
            int e8 = tid + ch * 128;
            int nr = e8 >> 2, kc = (e8 & 3) * 8;
            STAGE_CP(Bt + (long long)(nBase + nr) * ldb + kb + kc, &Bs[buf][nr][kc]);
        }
    };

    stage(0, 0);
    STAGE_WAIT();
    __syncthreads();

    int cur = 0;
    for (int kb = 0; kb < K; kb += 32) {
        int nxt = cur ^ 1;
        if (kb + 32 < K) {
            stage(nxt, kb + 32);                          // overlap with math below
            if (kb + 64 < K) {                            // prefetch one more step ahead
                __builtin_prefetch((const void*)(A + (long long)(mBase + (tid >> 1)) * lda + kb + 64), 0, 0);
                __builtin_prefetch((const void*)(Bt + (long long)(nBase + (tid & (TN - 1))) * ldb + kb + 64), 0, 0);
            }
        }
        // ---- fragment loads (16-bit A 16x32 layout; B symmetric via transpose) ----
        FragU a[2], b[JT];
#pragma unroll
        for (int i = 0; i < 2; ++i) {
            int m = wRow + i * 16 + lm;
#pragma unroll
            for (int e2 = 0; e2 < 8; ++e2) {
                int kk = (e2 < 4) ? (half * 8 + 2 * e2) : (8 + half * 8 + 2 * e2);
                a[i].u[e2] = *(const unsigned int*)&As[cur][m][kk];
            }
        }
#pragma unroll
        for (int j = 0; j < JT; ++j) {
            int n = wCol + j * 16 + lm;
#pragma unroll
            for (int e2 = 0; e2 < 8; ++e2) {
                int kk = (e2 < 4) ? (half * 8 + 2 * e2) : (8 + half * 8 + 2 * e2);
                b[j].u[e2] = *(const unsigned int*)&Bs[cur][n][kk];
            }
        }
#pragma unroll
        for (int i = 0; i < 2; ++i)
#pragma unroll
            for (int j = 0; j < JT; ++j)
                c[i][j].v = __builtin_amdgcn_wmma_f32_16x16x32_bf16(
                    false, a[i].v, false, b[j].v, (short)0, c[i][j].v, false, false);
        STAGE_WAIT();
        __syncthreads();
        cur = nxt;
    }

    // ---- epilogue: C 16x16 f32 layout: VGPR v -> M = v + half*8, N = lane&15 ----
#pragma unroll
    for (int i = 0; i < 2; ++i)
#pragma unroll
        for (int j = 0; j < JT; ++j) {
            int n = nBase + wCol + j * 16 + lm;
#pragma unroll
            for (int e = 0; e < 8; ++e) {
                int m = mBase + wRow + i * 16 + e + half * 8;
                float val = c[i][j].f[e];
                if (rowScale) val *= rowScale[m];
                long long idx = cOff + (long long)m * ldc + n;
                if (Cbf) Cbf[idx] = f2bf(val);
                else     Cf[idx]  = val;
            }
        }
}

// ---------------- elementwise / small kernels ----------------
__global__ void wconv_kernel(const float* __restrict__ w, unsigned short* __restrict__ o, int n)
{
    for (int i = blockIdx.x * blockDim.x + threadIdx.x; i < n; i += gridDim.x * blockDim.x)
        o[i] = f2bf(w[i]);
}

// transposed bf16 weight convert: dst[z, n, k] = src[z, k, n] * (rowScale ? rowScale[k] : 1)
__global__ void wconvT_kernel(const float* __restrict__ src, unsigned short* __restrict__ dst,
                              int K, int N, int total, const float* __restrict__ rowScale)
{
    int kn = K * N;
    for (int i = blockIdx.x * blockDim.x + threadIdx.x; i < total; i += gridDim.x * blockDim.x) {
        int z = i / kn, r = i - z * kn;
        int n = r / K, k = r - n * K;
        float v = src[(long long)z * kn + (long long)k * N + n];
        if (rowScale) v *= rowScale[k];
        dst[i] = f2bf(v);
    }
}

__global__ void add_kernel(const float* __restrict__ a, const float* __restrict__ b,
                           float* __restrict__ o, int n)
{
    for (int i = blockIdx.x * blockDim.x + threadIdx.x; i < n; i += gridDim.x * blockDim.x)
        o[i] = a[i] + b[i];
}

__global__ void silu_mul_kernel(const float* __restrict__ g, const float* __restrict__ u,
                                unsigned short* __restrict__ o, int n)
{
    for (int i = blockIdx.x * blockDim.x + threadIdx.x; i < n; i += gridDim.x * blockDim.x) {
        float gv = g[i];
        float s  = gv / (1.f + __expf(-gv));
        o[i] = f2bf(s * u[i]);
    }
}

__global__ void bias_relu_kernel(const float* __restrict__ in, const float* __restrict__ bias,
                                 int C, unsigned short* __restrict__ o, int n)
{
    for (int i = blockIdx.x * blockDim.x + threadIdx.x; i < n; i += gridDim.x * blockDim.x) {
        float v = in[i] + bias[i % C];
        o[i] = f2bf(fmaxf(v, 0.f));
    }
}

// out[b, i*64 + j*8 + s] = in[b, j*64 + i*8 + s] (*mask) + res[...]
__global__ void headmix_kernel(const float* __restrict__ in, const float* __restrict__ res,
                               float* __restrict__ o, int applyMask)
{
    int i = blockIdx.x * blockDim.x + threadIdx.x;
    if (i >= B_ * TD_) return;
    int b = i >> 9, d = i & 511;
    int oh = d >> 6, rest = d & 63, jh = rest >> 3, s = rest & 7;
    float v = in[b * TD_ + jh * 64 + oh * 8 + s];
    if (applyMask && oh < NU_ && jh >= NU_) v = 0.f;
    o[i] = v + res[i];
}

// per-head RMS norm on rows of 64; grid = numRows(=B*N), block = 64
__global__ void rms_head_kernel(const float* __restrict__ in, const float* __restrict__ w,
                                float* __restrict__ outF, unsigned short* __restrict__ outB)
{
    __shared__ float red[64];
    int row = blockIdx.x, d = threadIdx.x;
    float v = in[row * 64 + d];
    red[d] = v * v; __syncthreads();
    for (int s = 32; s > 0; s >>= 1) { if (d < s) red[d] += red[d + s]; __syncthreads(); }
    float inv = rsqrtf(red[0] / 64.f + EPS_);
    float o = v * inv * w[d];
    if (outF) outF[row * 64 + d] = o;
    if (outB) outB[row * 64 + d] = f2bf(o);
}

// flat RMS norm, rowLen cols; grid = rows, block = 256
__global__ void rms_flat_kernel(const float* __restrict__ in, const float* __restrict__ w,
                                float* __restrict__ outF, unsigned short* __restrict__ outB,
                                int rowLen)
{
    __shared__ float red[256];
    int row = blockIdx.x, t = threadIdx.x;
    float acc = 0.f;
    for (int c = t; c < rowLen; c += 256) { float v = in[row * rowLen + c]; acc += v * v; }
    red[t] = acc; __syncthreads();
    for (int s = 128; s > 0; s >>= 1) { if (t < s) red[t] += red[t + s]; __syncthreads(); }
    float inv = rsqrtf(red[0] / (float)rowLen + EPS_);
    for (int c = t; c < rowLen; c += 256) {
        float o = in[row * rowLen + c] * inv * w[c];
        if (outF) outF[row * rowLen + c] = o;
        if (outB) outB[row * rowLen + c] = f2bf(o);
    }
}

// inverse RMS of each bf16 seq row (layer independent); grid = BT, block = 256
__global__ void rowinvrms_kernel(const unsigned short* __restrict__ seq, float* __restrict__ srow)
{
    __shared__ float red[256];
    int row = blockIdx.x, t = threadIdx.x;
    float acc = 0.f;
    for (int c = t; c < TD_; c += 256) { float v = bf2f(seq[(long long)row * TD_ + c]); acc += v * v; }
    red[t] = acc; __syncthreads();
    for (int s = 128; s > 0; s >>= 1) { if (t < s) red[t] += red[t + s]; __syncthreads(); }
    if (t == 0) srow[row] = rsqrtf(red[0] / (float)TD_ + EPS_);
}

// gather + concat sequence embeddings -> bf16 (BT, 96)
__global__ void seqembed_kernel(const float* __restrict__ e0, const float* __restrict__ e1,
                                const int* __restrict__ sf0, const int* __restrict__ sf1,
                                unsigned short* __restrict__ s)
{
    int total = BT_ * 96;
    for (int i = blockIdx.x * blockDim.x + threadIdx.x; i < total; i += gridDim.x * blockDim.x) {
        int bt = i / 96, c = i - bt * 96;
        float v = (c < 64) ? e0[(long long)sf0[bt] * 64 + c]
                           : e1[(long long)sf1[bt] * 32 + (c - 64)];
        s[i] = f2bf(v);
    }
}

// UI feature embedding + per-head projection -> x (B, N, D) f32
__global__ void embedx_kernel(const float* e0, const float* e1, const float* e2,
                              const float* e3, const float* e4, const float* e5,
                              const int* n0, const int* n1, const int* n2,
                              const int* n3, const int* n4, const int* n5,
                              const float* __restrict__ uproj, const float* __restrict__ iproj,
                              float* __restrict__ x)
{
    __shared__ float ev[32];
    int b = blockIdx.x, h = blockIdx.y, d = threadIdx.x;
    if (d < 32) {
        const float* tb; long long r; int off, dim;
        switch (h) {
            case 0: tb = e0; r = n0[b]; off = 0;  dim = 64; break;
            case 1: tb = e0; r = n0[b]; off = 32; dim = 64; break;
            case 2: tb = e1; r = n1[b]; off = 0;  dim = 32; break;
            case 3: tb = e2; r = n2[b]; off = 0;  dim = 32; break;
            case 4: tb = e3; r = n3[b]; off = 0;  dim = 64; break;
            case 5: tb = e3; r = n3[b]; off = 32; dim = 64; break;
            case 6: tb = e4; r = n4[b]; off = 0;  dim = 32; break;
            default: tb = e5; r = n5[b]; off = 0; dim = 32; break;
        }
        ev[d] = tb[r * dim + off + d];
    }
    __syncthreads();
    const float* proj = (h < NU_) ? (uproj + h * 32 * 64) : (iproj + (h - NU_) * 32 * 64);
    float acc = 0.f;
#pragma unroll 8
    for (int s2 = 0; s2 < 32; ++s2) acc += ev[s2] * proj[s2 * 64 + d];
    x[(long long)b * TD_ + h * 64 + d] = acc;
}

// one block per (b, head): scores -> softmax -> ctx ; k/v are bf16 (BT, 512)
__global__ __launch_bounds__(256)
void attention_kernel(const float* __restrict__ qh,
                      const unsigned short* __restrict__ k,
                      const unsigned short* __restrict__ v,
                      const unsigned char* __restrict__ mask,
                      float* __restrict__ ctx)
{
    __shared__ float qs[64];
    __shared__ float ls[256];
    __shared__ float red[256];
    int b = blockIdx.x, n = blockIdx.y, t = threadIdx.x;
    if (t < 64) qs[t] = qh[(long long)b * TD_ + n * 64 + t];
    __syncthreads();

    float sc = -1e30f;
    if (t < T_) {
        const unsigned short* kp = k + ((long long)(b * T_ + t) * TD_ + n * 64);
        float acc = 0.f;
#pragma unroll 8
        for (int d = 0; d < 64; ++d) acc += qs[d] * bf2f(kp[d]);
        sc = acc * 0.125f;                       // 1/sqrt(64)
        if (!mask[b * T_ + t]) sc = -1e9f;
    }
    ls[t] = sc;
    red[t] = sc; __syncthreads();
    for (int s = 128; s > 0; s >>= 1) { if (t < s) red[t] = fmaxf(red[t], red[t + s]); __syncthreads(); }
    float mx = red[0]; __syncthreads();

    float p = (t < T_) ? __expf(ls[t] - mx) : 0.f;
    ls[t] = p;
    red[t] = p; __syncthreads();
    for (int s = 128; s > 0; s >>= 1) { if (t < s) red[t] += red[t + s]; __syncthreads(); }
    float inv = 1.f / red[0]; __syncthreads();

    if (t < 64) {
        float acc = 0.f;
        const unsigned short* vp = v + ((long long)(b * T_) * TD_ + n * 64 + t);
        for (int tt = 0; tt < T_; ++tt) acc += ls[tt] * bf2f(vp[(long long)tt * TD_]);
        ctx[(long long)b * TD_ + n * 64 + t] = acc * inv;
    }
}

// final 128-dim reduction per row: out[b] = h2 . w3 + b3
__global__ void logit_kernel(const unsigned short* __restrict__ h2, const float* __restrict__ w3,
                             const float* __restrict__ b3, float* __restrict__ out)
{
    __shared__ float red[128];
    int b = blockIdx.x, d = threadIdx.x;
    red[d] = bf2f(h2[b * 128 + d]) * w3[d]; __syncthreads();
    for (int s = 64; s > 0; s >>= 1) { if (d < s) red[d] += red[d + s]; __syncthreads(); }
    if (d == 0) out[b] = red[0] + b3[0];
}

// ---------------- host side ----------------
// layer param offsets (JAX dict-key-sorted flattening)
enum { LP_ca_qn = 0, LP_ca_sn, LP_ffn_n, LP_of_n1, LP_of_n2, LP_of_w1, LP_of_w2, LP_of_w3,
       LP_qm_n1, LP_qm_n2, LP_qm_w1, LP_qm_w2, LP_qm_w3, LP_wd, LP_wg, LP_wk, LP_wo,
       LP_wq, LP_wu, LP_wv };

static inline void gemm(hipStream_t st,
                        const unsigned short* A, int lda, long long sAz, const float* rs,
                        const unsigned short* Bt, int ldb, long long sBz,
                        float* Cf, unsigned short* Cbf, int ldc, long long sCz,
                        int M, int N, int K, int batch)
{
    if ((N & 127) == 0) {
        dim3 g(N / 128, M / 64, batch);
        gemm_bf16_kernel<128><<<g, dim3(128), 0, st>>>(A, lda, sAz, rs, Bt, ldb, sBz, Cf, Cbf, ldc, sCz, K);
    } else {
        dim3 g(N / 64, M / 64, batch);
        gemm_bf16_kernel<64><<<g, dim3(128), 0, st>>>(A, lda, sAz, rs, Bt, ldb, sBz, Cf, Cbf, ldc, sCz, K);
    }
}

extern "C" void kernel_launch(void* const* d_in, const int* in_sizes, int n_in,
                              void* d_out, int out_size, void* d_ws, size_t ws_size,
                              hipStream_t stream)
{
    (void)in_sizes; (void)n_in; (void)out_size; (void)ws_size;
    // ---- input index map (params pytree, dict keys sorted) ----
    const float* emb[6];      for (int i = 0; i < 6; ++i) emb[i] = (const float*)d_in[i];
    const float* final_norm = (const float*)d_in[6];
    const float* item_proj  = (const float*)d_in[7];
    const float* LPP[L_][20];
    for (int l = 0; l < L_; ++l)
        for (int j = 0; j < 20; ++j) LPP[l][j] = (const float*)d_in[8 + l * 20 + j];
    const float* seq_emb0 = (const float*)d_in[88];
    const float* seq_emb1 = (const float*)d_in[89];
    const float* seq_proj = (const float*)d_in[90];
    const float* TSK[2][6];
    for (int t = 0; t < 2; ++t)
        for (int j = 0; j < 6; ++j) TSK[t][j] = (const float*)d_in[91 + t * 6 + j]; // b1,b2,b3,w1,w2,w3
    const float* user_proj = (const float*)d_in[103];
    const int* nsf[6];        for (int i = 0; i < 6; ++i) nsf[i] = (const int*)d_in[104 + i];
    const int* sf0 = (const int*)d_in[110];
    const int* sf1 = (const int*)d_in[111];
    const unsigned char* seq_mask = (const unsigned char*)d_in[112];
    float* out = (float*)d_out;

    // ---- workspace allocator ----
    size_t off = 0;
    char* base = (char*)d_ws;
    auto alloc = [&](size_t bytes) -> void* {
        void* p = base + off;
        off += (bytes + 255) & ~(size_t)255;
        return p;
    };
    unsigned short* seq_bf = (unsigned short*)alloc((size_t)BT_ * TD_ * 2);
    unsigned short* kbuf   = (unsigned short*)alloc((size_t)BT_ * TD_ * 2);
    unsigned short* vbuf   = (unsigned short*)alloc((size_t)BT_ * TD_ * 2);
    unsigned short* sbuf   = (unsigned short*)alloc((size_t)BT_ * 96 * 2);
    float* srow            = (float*)alloc((size_t)BT_ * 4);
    float* xbuf            = (float*)alloc((size_t)B_ * TD_ * 4);
    float* tA  = (float*)alloc((size_t)B_ * TD_ * 4);
    float* tB  = (float*)alloc((size_t)B_ * TD_ * 4);
    float* tC  = (float*)alloc((size_t)B_ * TD_ * 4);
    float* tD  = (float*)alloc((size_t)B_ * TD_ * 4);
    float* tE  = (float*)alloc((size_t)B_ * TD_ * 4);
    float* tF  = (float*)alloc((size_t)B_ * TD_ * 4);
    float* tI  = (float*)alloc((size_t)B_ * TD_ * 4);
    float* qhB = (float*)alloc((size_t)B_ * TD_ * 4);
    float* ctx = (float*)alloc((size_t)B_ * TD_ * 4);
    float* gh  = (float*)alloc((size_t)B_ * N_ * H_ * 4);
    float* uh  = (float*)alloc((size_t)B_ * N_ * H_ * 4);
    float* tG  = (float*)alloc((size_t)B_ * SH_ * 4);
    float* tH  = (float*)alloc((size_t)B_ * SH_ * 4);
    unsigned short* abf   = (unsigned short*)alloc((size_t)B_ * TD_ * 2);
    unsigned short* hbf   = (unsigned short*)alloc((size_t)B_ * N_ * H_ * 2);
    unsigned short* hbf2  = (unsigned short*)alloc((size_t)B_ * SH_ * 2);
    unsigned short* sprojT = (unsigned short*)alloc((size_t)96 * TD_ * 2);
    unsigned short *wq_b[L_], *wk_b[L_], *wv_b[L_], *wo_b[L_], *wg_b[L_], *wu_b[L_], *wd_b[L_];
    unsigned short *qw1_b[L_], *qw2_b[L_], *qw3_b[L_], *ow1_b[L_], *ow2_b[L_], *ow3_b[L_];
    for (int l = 0; l < L_; ++l) {
        wq_b[l]  = (unsigned short*)alloc((size_t)TD_ * TD_ * 2);
        wk_b[l]  = (unsigned short*)alloc((size_t)TD_ * TD_ * 2);
        wv_b[l]  = (unsigned short*)alloc((size_t)TD_ * TD_ * 2);
        wo_b[l]  = (unsigned short*)alloc((size_t)TD_ * TD_ * 2);
        wg_b[l]  = (unsigned short*)alloc((size_t)TD_ * SH_ * 2);
        wu_b[l]  = (unsigned short*)alloc((size_t)TD_ * SH_ * 2);
        wd_b[l]  = (unsigned short*)alloc((size_t)SH_ * TD_ * 2);
        qw1_b[l] = (unsigned short*)alloc((size_t)N_ * D_ * H_ * 2);
        qw3_b[l] = (unsigned short*)alloc((size_t)N_ * D_ * H_ * 2);
        qw2_b[l] = (unsigned short*)alloc((size_t)N_ * H_ * D_ * 2);
        ow1_b[l] = (unsigned short*)alloc((size_t)N_ * D_ * H_ * 2);
        ow3_b[l] = (unsigned short*)alloc((size_t)N_ * D_ * H_ * 2);
        ow2_b[l] = (unsigned short*)alloc((size_t)N_ * H_ * D_ * 2);
    }
    unsigned short *tw1_b[2], *tw2_b[2];
    for (int t = 0; t < 2; ++t) {
        tw1_b[t] = (unsigned short*)alloc((size_t)TD_ * 256 * 2);
        tw2_b[t] = (unsigned short*)alloc((size_t)256 * 128 * 2);
    }

    // transposed bf16 conversion: src (z,K,N) -> dst (z,N,K); rowScale indexed by k
    auto convT = [&](const float* src, unsigned short* dst, int K, int N, int batch,
                     const float* rs) {
        int total = batch * K * N;
        wconvT_kernel<<<dim3((total + 255) / 256), dim3(256), 0, stream>>>(src, dst, K, N, total, rs);
    };
    auto vadd = [&](const float* a, const float* b, float* o, int n) {
        add_kernel<<<dim3((n + 255) / 256), dim3(256), 0, stream>>>(a, b, o, n);
    };

    // ---- weight conversions (bf16, transposed; fold ca_sn into wk/wv rows) ----
    convT(seq_proj, sprojT, 96, TD_, 1, nullptr);
    for (int l = 0; l < L_; ++l) {
        convT(LPP[l][LP_wq], wq_b[l], TD_, TD_, 1, nullptr);
        convT(LPP[l][LP_wk], wk_b[l], TD_, TD_, 1, LPP[l][LP_ca_sn]);
        convT(LPP[l][LP_wv], wv_b[l], TD_, TD_, 1, LPP[l][LP_ca_sn]);
        convT(LPP[l][LP_wo], wo_b[l], TD_, TD_, 1, nullptr);
        convT(LPP[l][LP_wg], wg_b[l], TD_, SH_, 1, nullptr);
        convT(LPP[l][LP_wu], wu_b[l], TD_, SH_, 1, nullptr);
        convT(LPP[l][LP_wd], wd_b[l], SH_, TD_, 1, nullptr);
        convT(LPP[l][LP_qm_w1], qw1_b[l], D_, H_, N_, nullptr);
        convT(LPP[l][LP_qm_w3], qw3_b[l], D_, H_, N_, nullptr);
        convT(LPP[l][LP_qm_w2], qw2_b[l], H_, D_, N_, nullptr);
        convT(LPP[l][LP_of_w1], ow1_b[l], D_, H_, N_, nullptr);
        convT(LPP[l][LP_of_w3], ow3_b[l], D_, H_, N_, nullptr);
        convT(LPP[l][LP_of_w2], ow2_b[l], H_, D_, N_, nullptr);
    }
    for (int t = 0; t < 2; ++t) {
        convT(TSK[t][3], tw1_b[t], TD_, 256, 1, nullptr);
        convT(TSK[t][4], tw2_b[t], 256, 128, 1, nullptr);
    }

    // ---- embeddings ----
    seqembed_kernel<<<dim3(4096), dim3(256), 0, stream>>>(seq_emb0, seq_emb1, sf0, sf1, sbuf);
    embedx_kernel<<<dim3(B_, N_), dim3(64), 0, stream>>>(
        emb[0], emb[1], emb[2], emb[3], emb[4], emb[5],
        nsf[0], nsf[1], nsf[2], nsf[3], nsf[4], nsf[5], user_proj, item_proj, xbuf);

    // ---- seq = s @ seq_proj (bf16 out) + row inv-rms ----
    gemm(stream, sbuf, 96, 0, nullptr, sprojT, 96, 0, nullptr, seq_bf, TD_, 0, BT_, TD_, 96, 1);
    rowinvrms_kernel<<<dim3(BT_), dim3(256), 0, stream>>>(seq_bf, srow);

    const int nBD = B_ * TD_;
    for (int l = 0; l < L_; ++l) {
        // --- UIQueryMixer ---
        rms_head_kernel<<<dim3(B_ * N_), dim3(64), 0, stream>>>(xbuf, LPP[l][LP_qm_n1], tA, nullptr);
        headmix_kernel<<<dim3((nBD + 255) / 256), dim3(256), 0, stream>>>(tA, xbuf, tB, 1);
        rms_head_kernel<<<dim3(B_ * N_), dim3(64), 0, stream>>>(tB, LPP[l][LP_qm_n2], nullptr, abf);
        gemm(stream, abf, TD_, 64, nullptr, qw1_b[l], D_, (long long)D_ * H_,
             gh, nullptr, N_ * H_, H_, B_, H_, D_, N_);
        gemm(stream, abf, TD_, 64, nullptr, qw3_b[l], D_, (long long)D_ * H_,
             uh, nullptr, N_ * H_, H_, B_, H_, D_, N_);
        silu_mul_kernel<<<dim3((B_ * N_ * H_ + 255) / 256), dim3(256), 0, stream>>>(gh, uh, hbf, B_ * N_ * H_);
        gemm(stream, hbf, N_ * H_, H_, nullptr, qw2_b[l], H_, (long long)H_ * D_,
             tC, nullptr, TD_, D_, B_, D_, H_, N_);
        vadd(tC, tB, tD, nBD);                                   // q (x-side)
        // --- CrossAttention ---
        rms_head_kernel<<<dim3(B_ * N_), dim3(64), 0, stream>>>(tD, LPP[l][LP_ca_qn], nullptr, abf);
        gemm(stream, abf, TD_, 0, nullptr, wq_b[l], TD_, 0, qhB, nullptr, TD_, 0, B_, TD_, TD_, 1);
        gemm(stream, seq_bf, TD_, 0, srow, wk_b[l], TD_, 0, nullptr, kbuf, TD_, 0, BT_, TD_, TD_, 1);
        gemm(stream, seq_bf, TD_, 0, srow, wv_b[l], TD_, 0, nullptr, vbuf, TD_, 0, BT_, TD_, TD_, 1);
        attention_kernel<<<dim3(B_, N_), dim3(256), 0, stream>>>(qhB, kbuf, vbuf, seq_mask, ctx);
        wconv_kernel<<<dim3((nBD + 255) / 256), dim3(256), 0, stream>>>(ctx, abf, nBD);
        gemm(stream, abf, TD_, 0, nullptr, wo_b[l], TD_, 0, tE, nullptr, TD_, 0, B_, TD_, TD_, 1);
        vadd(tE, tD, tF, nBD);                                   // z
        // --- sequence-side FFN ---
        rms_flat_kernel<<<dim3(B_), dim3(256), 0, stream>>>(tF, LPP[l][LP_ffn_n], nullptr, abf, TD_);
        gemm(stream, abf, TD_, 0, nullptr, wg_b[l], TD_, 0, tG, nullptr, SH_, 0, B_, SH_, TD_, 1);
        gemm(stream, abf, TD_, 0, nullptr, wu_b[l], TD_, 0, tH, nullptr, SH_, 0, B_, SH_, TD_, 1);
        silu_mul_kernel<<<dim3((B_ * SH_ + 255) / 256), dim3(256), 0, stream>>>(tG, tH, hbf2, B_ * SH_);
        gemm(stream, hbf2, SH_, 0, nullptr, wd_b[l], SH_, 0, tE, nullptr, TD_, 0, B_, TD_, SH_, 1);
        vadd(tE, tF, tI, nBD);                                   // z2
        // --- OutputFusion ---
        rms_head_kernel<<<dim3(B_ * N_), dim3(64), 0, stream>>>(tI, LPP[l][LP_of_n1], tA, nullptr);
        headmix_kernel<<<dim3((nBD + 255) / 256), dim3(256), 0, stream>>>(tA, tI, tB, 0);
        rms_head_kernel<<<dim3(B_ * N_), dim3(64), 0, stream>>>(tB, LPP[l][LP_of_n2], nullptr, abf);
        gemm(stream, abf, TD_, 64, nullptr, ow1_b[l], D_, (long long)D_ * H_,
             gh, nullptr, N_ * H_, H_, B_, H_, D_, N_);
        gemm(stream, abf, TD_, 64, nullptr, ow3_b[l], D_, (long long)D_ * H_,
             uh, nullptr, N_ * H_, H_, B_, H_, D_, N_);
        silu_mul_kernel<<<dim3((B_ * N_ * H_ + 255) / 256), dim3(256), 0, stream>>>(gh, uh, hbf, B_ * N_ * H_);
        gemm(stream, hbf, N_ * H_, H_, nullptr, ow2_b[l], H_, (long long)H_ * D_,
             tC, nullptr, TD_, D_, B_, D_, H_, N_);
        vadd(tC, tB, xbuf, nBD);                                 // next x
    }

    // ---- final norm + task heads ----
    rms_head_kernel<<<dim3(B_ * N_), dim3(64), 0, stream>>>(xbuf, final_norm, nullptr, abf);
    for (int t = 0; t < 2; ++t) {
        gemm(stream, abf, TD_, 0, nullptr, tw1_b[t], TD_, 0, tG, nullptr, 256, 0, B_, 256, TD_, 1);
        bias_relu_kernel<<<dim3((B_ * 256 + 255) / 256), dim3(256), 0, stream>>>(tG, TSK[t][0], 256, hbf, B_ * 256);
        gemm(stream, hbf, 256, 0, nullptr, tw2_b[t], 256, 0, tH, nullptr, 128, 0, B_, 128, 256, 1);
        bias_relu_kernel<<<dim3((B_ * 128 + 255) / 256), dim3(256), 0, stream>>>(tH, TSK[t][1], 128, hbf2, B_ * 128);
        logit_kernel<<<dim3(B_), dim3(128), 0, stream>>>(hbf2, TSK[t][5], TSK[t][2], out + t * B_);
    }
}
